// textPCNN_73272142070336
// MI455X (gfx1250) — matrix operand order repaired and checked by reading, not compiled, not run on hardware
//
#include <hip/hip_runtime.h>
#include <hip/hip_bf16.h>

typedef __attribute__((ext_vector_type(16))) __bf16        v16bf;
typedef __attribute__((ext_vector_type(8)))  float         v8f;
typedef __attribute__((ext_vector_type(4)))  unsigned int  v4u;

#define N_SAMP 1024
#define SEQ_L  128
#define WD_    300
#define PD_    50
#define D_     400
#define NF_    512
#define C_     53
#define LP_    126
#define KTOT   1200
#define KPAD   1216
#define NEGV   (-1e30f)

// LDS carve (bytes)
#define SX_ELEMS   52032            // >= 127*400+1215+1, 16B-multiple
#define SX_BYTES   (SX_ELEMS * 2)   // 104064
#define WB_OFF     SX_BYTES                 // 4 M-groups * 3 segs * 512 floats
#define PB_OFF     (WB_OFF + 4*3*512*4)     // 128640 : feat[1536]
#define PT_OFF     (PB_OFF + 1536*4)        // 134784 : partial[4][64]
#define SMEM_BYTES (PT_OFF + 4*64*4)        // 135808

union Frag { v16bf bf; v4u u2[2]; };

__device__ __forceinline__ unsigned short f2bf(float x) {
  unsigned u = __float_as_uint(x);
  u += 0x7FFFu + ((u >> 16) & 1u);
  return (unsigned short)(u >> 16);
}

// A-frag from LDS: 16-bit A layout, lane holds K runs [ka,ka+8) and
// [ka+16,ka+24) -> two ds_load_b128.
#define LOADA(FR, AP, KOFF)                                        \
  do {                                                             \
    (FR).u2[0] = *(const v4u*)((AP) + (KOFF));                     \
    (FR).u2[1] = *(const v4u*)((AP) + (KOFF) + 16);                \
  } while (0)

// B-frag from global: lane holds K runs [kb,kb+8) and [kb+16,kb+24) of
// column f -> two global_load_b128.
#define LOADB(FR, BP, KOFF)                                        \
  do {                                                             \
    (FR).u2[0] = *(const v4u*)((BP) + (KOFF));                     \
    (FR).u2[1] = *(const v4u*)((BP) + (KOFF) + 16);                \
  } while (0)

#define WMMA(A, B, CACC)                                           \
  __builtin_amdgcn_wmma_f32_16x16x32_bf16(false, (A).bf, false,    \
                                          (B).bf, (short)0, (CACC), false, false)

// Kernel 0: conv_w (512 x 1200 f32) -> bf16 (512 x 1216), zero-padded K tail.
__global__ __launch_bounds__(256) void convw_bf16(const float* __restrict__ w,
                                                  unsigned short* __restrict__ o) {
  int idx = blockIdx.x * 256 + threadIdx.x;
  if (idx >= NF_ * KPAD) return;
  int f = idx / KPAD;
  int k = idx - f * KPAD;
  float v = (k < KTOT) ? w[f * KTOT + k] : 0.0f;
  o[idx] = f2bf(v);
}

// Kernel 1: fused gather -> im2col GEMM (WMMA bf16, 2Mx4N register tile) ->
//           tanh -> piecewise max pool -> output linear.
//           One block per sample, 256 threads (8 waves).
__global__ __launch_bounds__(256) void pcnn_fused(
    const int* __restrict__ tokens, const int* __restrict__ pf1,
    const int* __restrict__ pf2, const int* __restrict__ epos,
    const float* __restrict__ wordvec, const float* __restrict__ pf1e,
    const float* __restrict__ pf2e, const unsigned short* __restrict__ gw,
    const float* __restrict__ conv_b, const float* __restrict__ out_w,
    const float* __restrict__ out_b, float* __restrict__ out) {
  extern __shared__ char smem[];
  unsigned short* sx   = (unsigned short*)smem;        // bf16 x: 128 rows x 400
  float*          wbuf = (float*)(smem + WB_OFF);      // [mg][seg][f]
  float*          pbuf = (float*)(smem + PB_OFF);      // feat[f*3+seg]
  float*          part = (float*)(smem + PT_OFF);      // [part][64]

  const int n   = blockIdx.x;
  const int tid = threadIdx.x;

  // ---- Phase 1: embedding gather -> LDS bf16 ----
  for (int i = tid; i < SEQ_L * D_; i += 256) {
    int l = i / D_;
    int d = i - l * D_;
    float v;
    if (d < WD_) {
      v = wordvec[(long)tokens[n * SEQ_L + l] * WD_ + d];
    } else if (d < WD_ + PD_) {
      v = pf1e[pf1[n * SEQ_L + l] * PD_ + (d - WD_)];
    } else {
      v = pf2e[pf2[n * SEQ_L + l] * PD_ + (d - WD_ - PD_)];
    }
    sx[i] = f2bf(v);
  }
  for (int i = SEQ_L * D_ + tid; i < SX_ELEMS; i += 256) sx[i] = 0;
  __syncthreads();

  const int p1 = epos[2 * n + 0];
  const int p2 = epos[2 * n + 1];

  const int wave  = tid >> 5;
  const int lane  = tid & 31;
  const int laneM = lane & 15;
  const int hi    = (lane >> 4) & 1;
  const int mg    = wave >> 1;           // M-group: rows [mg*32, mg*32+32)
  const int ng    = wave & 1;            // N-half:  f in [ng*256, ng*256+256)
  const int ka    = hi ? 8 : 0;          // 16-bit A layout K-run base
  const int kb    = hi ? 8 : 0;          // 16-bit B layout K-run base

  const unsigned short* arow0 = sx + (mg * 32 + 0  + laneM) * D_ + ka;
  const unsigned short* arow1 = sx + (mg * 32 + 16 + laneM) * D_ + ka;
  const unsigned short* bcol  = gw + (ng * 256 + laneM) * KPAD + kb;

  // ---- Phase 2: software-pipelined WMMA GEMM + fused tanh/pool epilogue ----
  for (int nc = 0; nc < 4; ++nc) {       // 4 chunks of 4 N-tiles (of wave's 16)
    v8f a00 = {}, a01 = {}, a02 = {}, a03 = {};   // M-tile 0 x 4 N-tiles
    v8f a10 = {}, a11 = {}, a12 = {}, a13 = {};   // M-tile 1 x 4 N-tiles
    const unsigned short* b0 = bcol + (nc * 64 + 0)  * KPAD;
    const unsigned short* b1 = bcol + (nc * 64 + 16) * KPAD;
    const unsigned short* b2 = bcol + (nc * 64 + 32) * KPAD;
    const unsigned short* b3 = bcol + (nc * 64 + 48) * KPAD;

    Frag fa0[2], fa1[2], f0[2], f1[2], f2[2], f3[2];
    // prologue: preload k-step 0 into buffer 0
    LOADA(fa0[0], arow0, 0);
    LOADA(fa1[0], arow1, 0);
    LOADB(f0[0], b0, 0);
    LOADB(f1[0], b1, 0);
    LOADB(f2[0], b2, 0);
    LOADB(f3[0], b3, 0);

#pragma unroll 2
    for (int ks = 0; ks < 37; ++ks) {    // steady state: load k+1, compute k
      const int cur = ks & 1;
      const int nxt = cur ^ 1;
      const int kn  = (ks + 1) * 32;
      LOADA(fa0[nxt], arow0, kn);
      LOADA(fa1[nxt], arow1, kn);
      LOADB(f0[nxt], b0, kn);
      LOADB(f1[nxt], b1, kn);
      LOADB(f2[nxt], b2, kn);
      LOADB(f3[nxt], b3, kn);
      a00 = WMMA(fa0[cur], f0[cur], a00);
      a01 = WMMA(fa0[cur], f1[cur], a01);
      a02 = WMMA(fa0[cur], f2[cur], a02);
      a03 = WMMA(fa0[cur], f3[cur], a03);
      a10 = WMMA(fa1[cur], f0[cur], a10);
      a11 = WMMA(fa1[cur], f1[cur], a11);
      a12 = WMMA(fa1[cur], f2[cur], a12);
      a13 = WMMA(fa1[cur], f3[cur], a13);
    }
    // tail k-step 37 (buffer 37&1 == 1)
    a00 = WMMA(fa0[1], f0[1], a00);
    a01 = WMMA(fa0[1], f1[1], a01);
    a02 = WMMA(fa0[1], f2[1], a02);
    a03 = WMMA(fa0[1], f3[1], a03);
    a10 = WMMA(fa1[1], f0[1], a10);
    a11 = WMMA(fa1[1], f1[1], a11);
    a12 = WMMA(fa1[1], f2[1], a12);
    a13 = WMMA(fa1[1], f3[1], a13);

    // Epilogue: tanh + bias, piecewise max over this wave's 32 rows.
    const v8f* am0[4] = {&a00, &a01, &a02, &a03};
    const v8f* am1[4] = {&a10, &a11, &a12, &a13};
#pragma unroll
    for (int j = 0; j < 4; ++j) {
      const int f = ng * 256 + (nc * 4 + j) * 16 + laneM;
      const float bias = conv_b[f];
      float s1 = NEGV, s2 = NEGV, s3 = NEGV;
#pragma unroll
      for (int m = 0; m < 2; ++m) {
        const v8f& ac = m ? *am1[j] : *am0[j];
        const int tb = mg * 32 + m * 16 + (hi ? 8 : 0);
#pragma unroll
        for (int r = 0; r < 8; ++r) {
          const int t = tb + r;
          const float v = (t < LP_) ? tanhf(ac[r] + bias) : NEGV;
          if (t <= p1)            s1 = fmaxf(s1, v);
          if (t >= p1 && t <= p2) s2 = fmaxf(s2, v);
          if (t >= p2)            s3 = fmaxf(s3, v);
        }
      }
      // merge lane L with lane L+16 (same f, other 8 rows of each M-tile)
      s1 = fmaxf(s1, __shfl_xor(s1, 16, 32));
      s2 = fmaxf(s2, __shfl_xor(s2, 16, 32));
      s3 = fmaxf(s3, __shfl_xor(s3, 16, 32));
      if (!hi) {
        float* wb = wbuf + mg * 1536;
        wb[0 * 512 + f] = s1;
        wb[1 * 512 + f] = s2;
        wb[2 * 512 + f] = s3;
      }
    }
  }
  __syncthreads();

  // ---- Phase 3: cross-M-group max reduce -> feat layout pbuf[f*3+seg] ----
  for (int k2 = tid; k2 < 3 * NF_; k2 += 256) {
    const int seg = k2 >> 9;
    const int f   = k2 & 511;
    float m = wbuf[seg * 512 + f];
#pragma unroll
    for (int g = 1; g < 4; ++g) m = fmaxf(m, wbuf[g * 1536 + seg * 512 + f]);
    pbuf[f * 3 + seg] = m;
  }
  __syncthreads();

  // ---- Phase 4: output linear (1536 x 53), split-K over 4 parts ----
  if (tid < 4 * C_) {
    const int pp = tid / C_;
    const int c  = tid - pp * C_;
    float a = 0.0f;
    const int j0 = pp * 384;
    for (int j = j0; j < j0 + 384; ++j) a += pbuf[j] * out_w[j * C_ + c];
    part[pp * 64 + c] = a;
  }
  __syncthreads();
  if (tid < C_) {
    float r = out_b[tid];
#pragma unroll
    for (int pp = 0; pp < 4; ++pp) r += part[pp * 64 + tid];
    out[n * C_ + tid] = r;
  }
}

extern "C" void kernel_launch(void* const* d_in, const int* in_sizes, int n_in,
                              void* d_out, int out_size, void* d_ws, size_t ws_size,
                              hipStream_t stream) {
  const int*   tokens  = (const int*)d_in[0];
  const int*   pf1     = (const int*)d_in[1];
  const int*   pf2     = (const int*)d_in[2];
  const int*   epos    = (const int*)d_in[3];
  const float* wordvec = (const float*)d_in[4];
  const float* pf1e    = (const float*)d_in[5];
  const float* pf2e    = (const float*)d_in[6];
  const float* conv_w  = (const float*)d_in[7];
  const float* conv_b  = (const float*)d_in[8];
  const float* out_w   = (const float*)d_in[9];
  const float* out_b   = (const float*)d_in[10];
  float*       out     = (float*)d_out;
  unsigned short* gw   = (unsigned short*)d_ws;  // bf16 conv_w, 512 x 1216

  const int tot = NF_ * KPAD;
  hipLaunchKernelGGL(convw_bf16, dim3((tot + 255) / 256), dim3(256), 0, stream,
                     conv_w, gw);
  hipLaunchKernelGGL(pcnn_fused, dim3(N_SAMP), dim3(256), SMEM_BYTES, stream,
                     tokens, pf1, pf2, epos, wordvec, pf1e, pf2e, gw, conv_b,
                     out_w, out_b, out);
}
